// GraphModel_50989851738514
// MI455X (gfx1250) — matrix operand order repaired and checked by reading, not compiled, not run on hardware
//
#include <hip/hip_runtime.h>
#include <math.h>

// ---------------- problem constants ----------------
constexpr int U_NODES = 8192;
constexpr int E_EDGES = 200000;
constexpr int DIM     = 768;
constexpr int DE      = 50;
constexpr int KE      = DIM + DE;   // 818
constexpr float BN_EPS = 1e-5f;

// ---------------- tiling ----------------
constexpr int BK = 32;              // K per wmma step
constexpr int NTHREADS = 256;       // 8 wave32 waves
constexpr int E_BM = 64, E_BN = 256;   // edge GEMM block tile; wave tile 32x64
constexpr int G_BM = 128, G_BN = 128;  // dense GEMM block tile; wave tile 32x64
constexpr int NSTEPS_E = (KE + BK - 1) / BK;  // 26 (24 fast + 2 tail)
constexpr int NSTEPS_G = DIM / BK;            // 24

typedef __bf16 v16bf __attribute__((ext_vector_type(16)));
typedef __bf16 v8bf  __attribute__((ext_vector_type(8)));
typedef float  v8f   __attribute__((ext_vector_type(8)));

// pack 8 f32 -> v8bf (compiler emits v_cvt_pk_bf16_f32 pairs); one ds_store_b128
__device__ inline v8bf pack8(float a0, float a1, float a2, float a3,
                             float b0, float b1, float b2, float b3) {
  v8bf r;
  r[0]=(__bf16)a0; r[1]=(__bf16)a1; r[2]=(__bf16)a2; r[3]=(__bf16)a3;
  r[4]=(__bf16)b0; r[5]=(__bf16)b1; r[6]=(__bf16)b2; r[7]=(__bf16)b3;
  return r;
}
__device__ inline v8bf pack8f4(float4 a, float4 b) {
  return pack8(a.x, a.y, a.z, a.w, b.x, b.y, b.z, b.w);
}

// A-fragment per CDNA5 wave32 layout (16-bit A 16x32):
// lane l: row M=l&15, khalf=l>>4; a[0..7]=K[khalf*8..+7], a[8..15]=K[16+khalf*8..+7]
__device__ inline v16bf make_a_frag(const __bf16* rowp, int khalf) {
  v8bf lo = *(const v8bf*)(rowp + khalf * 8);
  v8bf hi = *(const v8bf*)(rowp + 16 + khalf * 8);
  return __builtin_shufflevector(lo, hi, 0,1,2,3,4,5,6,7,8,9,10,11,12,13,14,15);
}

// wave computes a 32x64 tile: 2 A-frags x 4 B-frags = 8 WMMAs per K-step.
__device__ inline void compute_step(const __bf16 (*sA)[BK], const __bf16 (*sBt)[BK],
                                    int m_w, int n_w, int cl, int kh, v8f acc[2][4]) {
  v16bf a0 = make_a_frag(&sA[m_w + cl][0], kh);
  v16bf a1 = make_a_frag(&sA[m_w + 16 + cl][0], kh);
  #pragma unroll
  for (int t = 0; t < 4; ++t) {
    v16bf b = make_a_frag(&sBt[n_w + t * 16 + cl][0], kh);
    acc[0][t] = __builtin_amdgcn_wmma_f32_16x16x32_bf16(false, a0, false, b,
                                                        (short)0, acc[0][t], false, false);
    acc[1][t] = __builtin_amdgcn_wmma_f32_16x16x32_bf16(false, a1, false, b,
                                                        (short)0, acc[1][t], false, false);
  }
}

__device__ inline void atomic_add_f32(float* p, float v) {
  __hip_atomic_fetch_add(p, v, __ATOMIC_RELAXED, __HIP_MEMORY_SCOPE_AGENT);
}

// ---------------- utility kernels ----------------
__global__ void zero_f32_kernel(float* __restrict__ p, int n) {
  int i = blockIdx.x * blockDim.x + threadIdx.x;
  if (i < n) p[i] = 0.0f;
}

__global__ void build_idx_kernel(const int* __restrict__ uniq, const int* __restrict__ src,
                                 int* __restrict__ out, int E, int U) {
  int i = blockIdx.x * blockDim.x + threadIdx.x;
  if (i >= E) return;
  int v = src[i];
  int lo = 0, hi = U;
  while (lo < hi) { int mid = (lo + hi) >> 1; if (uniq[mid] < v) lo = mid + 1; else hi = mid; }
  out[i] = lo;
}

__global__ void bnstat_kernel(const float* __restrict__ colsum, const float* __restrict__ colsumsq,
                              const float* __restrict__ gamma, const float* __restrict__ beta,
                              float* __restrict__ scale, float* __restrict__ shift) {
  int c = blockIdx.x * blockDim.x + threadIdx.x;
  if (c >= DIM) return;
  const float inv_n = 1.0f / (float)U_NODES;
  float mean = colsum[c] * inv_n;
  float var  = colsumsq[c] * inv_n - mean * mean;   // biased, matches torch BN
  float sc   = gamma[c] * rsqrtf(var + BN_EPS);
  scale[c] = sc;
  shift[c] = beta[c] - mean * sc;
}

// ---------------- kernel 1: edge GEMM + bias + relu + scatter-add ----------------
__global__ __launch_bounds__(NTHREADS)
void edge_gemm_kernel(const float* __restrict__ tgt, const float* __restrict__ ef,
                      const float* __restrict__ We,  const float* __restrict__ be,
                      const int* __restrict__ idx,   float* __restrict__ agg) {
  __shared__ __align__(32) __bf16 sA[2][E_BM][BK];   //  8 KB
  __shared__ __align__(32) __bf16 sBt[2][E_BN][BK];  // 32 KB
  __shared__ int sIdx[E_BM];

  const int tid = threadIdx.x;
  const int e0 = blockIdx.x * E_BM;      // E % 64 == 0
  const int n0 = blockIdx.y * E_BN;
  if (tid < E_BM) sIdx[tid] = idx[e0 + tid];

  const int lane = tid & 31, w = tid >> 5;
  const int m_w = (w & 1) * 32, n_w = (w >> 1) * 64;
  const int cl = lane & 15, kh = lane >> 4;

  const int arow = tid >> 2, acs = (tid & 3) * 8;    // A: 64 rows x 4 threads of 8
  const size_t abase = (size_t)(e0 + arow) * DIM;
  const size_t ebase = (size_t)(e0 + arow) * DE;
  const int bn = tid;                                 // B: 1 thread per n-row of 32 k
  const size_t wbase = (size_t)(n0 + bn) * KE;

  auto stageA_fast = [&](int buf, int k0) {           // k0+32 <= 768
    const float4* p = (const float4*)(tgt + abase + k0 + acs);
    float4 v0 = p[0], v1 = p[1];
    *(v8bf*)(&sA[buf][arow][acs]) = pack8f4(v0, v1);  // one ds_store_b128
  };
  auto stageA_tail = [&](int buf, int k0) {           // concat boundary + zero pad
    float v[8];
    #pragma unroll
    for (int j = 0; j < 8; ++j) {
      int g = k0 + acs + j;
      v[j] = (g < DIM) ? tgt[abase + g] : (g < KE) ? ef[ebase + (g - DIM)] : 0.0f;
    }
    *(v8bf*)(&sA[buf][arow][acs]) = pack8(v[0],v[1],v[2],v[3],v[4],v[5],v[6],v[7]);
  };
  auto stageB_fast = [&](int buf, int k0) {           // k0+31 < 768 < 818
    const float2* p = (const float2*)(We + wbase + k0);  // even element offset
    #pragma unroll
    for (int q = 0; q < 4; ++q) {
      float2 a = p[4*q], b = p[4*q+1], c = p[4*q+2], d = p[4*q+3];
      *(v8bf*)(&sBt[buf][bn][8*q]) = pack8(a.x,a.y,b.x,b.y,c.x,c.y,d.x,d.y);
    }
  };
  auto stageB_tail = [&](int buf, int k0) {
    #pragma unroll
    for (int q = 0; q < 4; ++q) {
      float v[8];
      #pragma unroll
      for (int j = 0; j < 8; ++j) {
        int g = k0 + 8*q + j;
        v[j] = (g < KE) ? We[wbase + g] : 0.0f;
      }
      *(v8bf*)(&sBt[buf][bn][8*q]) = pack8(v[0],v[1],v[2],v[3],v[4],v[5],v[6],v[7]);
    }
  };

  v8f acc[2][4] = {};
  int buf = 0;
  stageA_fast(0, 0); stageB_fast(0, 0);
  __syncthreads();
  for (int s = 0; s < NSTEPS_E; ++s) {
    if (s + 1 < NSTEPS_E) {              // uniform branch: prefetch next tile
      int kn = (s + 1) * BK;
      if (kn + BK <= DIM) { stageA_fast(buf ^ 1, kn); stageB_fast(buf ^ 1, kn); }
      else                { stageA_tail(buf ^ 1, kn); stageB_tail(buf ^ 1, kn); }
    }
    compute_step(sA[buf], sBt[buf], m_w, n_w, cl, kh, acc);
    __syncthreads();
    buf ^= 1;
  }

  // epilogue: bias + relu + atomic scatter-sum (C layout: M = v + 8*kh, N = cl)
  #pragma unroll
  for (int i = 0; i < 2; ++i) {
    #pragma unroll
    for (int t = 0; t < 4; ++t) {
      int col = n0 + n_w + t * 16 + cl;
      float bias = be[col];
      #pragma unroll
      for (int v = 0; v < 8; ++v) {
        int r = m_w + i * 16 + kh * 8 + v;
        float val = acc[i][t][v] + bias;
        val = val > 0.0f ? val : 0.0f;
        atomic_add_f32(&agg[(size_t)sIdx[r] * DIM + col], val);
      }
    }
  }
}

// ---------------- kernel 2: h=(1+eps)*x+agg; h1 = h @ W1^T; BN partials ----------------
__global__ __launch_bounds__(NTHREADS)
void gemm1_kernel(const float* __restrict__ x, const float* __restrict__ agg,
                  const float* __restrict__ eps, const float* __restrict__ W1,
                  float* __restrict__ h1, float* __restrict__ colsum,
                  float* __restrict__ colsumsq) {
  __shared__ __align__(32) __bf16 sA[2][G_BM][BK];   // 16 KB
  __shared__ __align__(32) __bf16 sBt[2][G_BN][BK];  // 16 KB

  const int tid = threadIdx.x;
  const int u0 = blockIdx.x * G_BM;
  const int n0 = blockIdx.y * G_BN;
  const float ge = 1.0f + eps[0];

  const int lane = tid & 31, w = tid >> 5;
  const int m_w = (w & 3) * 32, n_w = (w >> 2) * 64;
  const int cl = lane & 15, kh = lane >> 4;

  const int arow = tid >> 1, acs = (tid & 1) * 16;   // A: 128 rows x 2 threads of 16
  const size_t abase = (size_t)(u0 + arow) * DIM;
  const int bn = tid >> 1, bcs = (tid & 1) * 16;     // B: 128 rows x 2 threads of 16
  const size_t bbase = (size_t)(n0 + bn) * DIM;

  auto stageA = [&](int buf, int k0) {               // fused GIN combine
    const float4* px = (const float4*)(x   + abase + k0 + acs);
    const float4* pa = (const float4*)(agg + abase + k0 + acs);
    #pragma unroll
    for (int q = 0; q < 2; ++q) {
      float4 x0 = px[2*q], x1 = px[2*q+1], a0 = pa[2*q], a1 = pa[2*q+1];
      *(v8bf*)(&sA[buf][arow][acs + 8*q]) =
        pack8(fmaf(ge,x0.x,a0.x), fmaf(ge,x0.y,a0.y), fmaf(ge,x0.z,a0.z), fmaf(ge,x0.w,a0.w),
              fmaf(ge,x1.x,a1.x), fmaf(ge,x1.y,a1.y), fmaf(ge,x1.z,a1.z), fmaf(ge,x1.w,a1.w));
    }
  };
  auto stageB = [&](int buf, int k0) {
    const float4* p = (const float4*)(W1 + bbase + k0 + bcs);
    #pragma unroll
    for (int q = 0; q < 2; ++q) {
      *(v8bf*)(&sBt[buf][bn][bcs + 8*q]) = pack8f4(p[2*q], p[2*q+1]);
    }
  };

  v8f acc[2][4] = {};
  int buf = 0;
  stageA(0, 0); stageB(0, 0);
  __syncthreads();
  for (int s = 0; s < NSTEPS_G; ++s) {
    if (s + 1 < NSTEPS_G) { stageA(buf ^ 1, (s + 1) * BK); stageB(buf ^ 1, (s + 1) * BK); }
    compute_step(sA[buf], sBt[buf], m_w, n_w, cl, kh, acc);
    __syncthreads();
    buf ^= 1;
  }

  // epilogue: store h1 + per-column sum / sumsq partials over the wave's 32 rows
  #pragma unroll
  for (int t = 0; t < 4; ++t) {
    int col = n0 + n_w + t * 16 + cl;
    float s = 0.0f, s2 = 0.0f;
    #pragma unroll
    for (int i = 0; i < 2; ++i) {
      #pragma unroll
      for (int v = 0; v < 8; ++v) {
        int r = m_w + i * 16 + kh * 8 + v;
        float val = acc[i][t][v];
        h1[(size_t)(u0 + r) * DIM + col] = val;
        s += val; s2 += val * val;
      }
    }
    s  += __shfl_xor(s, 16);
    s2 += __shfl_xor(s2, 16);
    if (kh == 0) { atomic_add_f32(&colsum[col], s); atomic_add_f32(&colsumsq[col], s2); }
  }
}

// ---------------- kernel 3: out = relu(BN(h1)) @ W2^T ----------------
__global__ __launch_bounds__(NTHREADS)
void gemm2_kernel(const float* __restrict__ h1, const float* __restrict__ scale,
                  const float* __restrict__ shift, const float* __restrict__ W2,
                  float* __restrict__ out) {
  __shared__ __align__(32) __bf16 sA[2][G_BM][BK];
  __shared__ __align__(32) __bf16 sBt[2][G_BN][BK];
  __shared__ float sScale[DIM];
  __shared__ float sShift[DIM];

  const int tid = threadIdx.x;
  const int u0 = blockIdx.x * G_BM;
  const int n0 = blockIdx.y * G_BN;

  for (int i = tid; i < DIM; i += NTHREADS) { sScale[i] = scale[i]; sShift[i] = shift[i]; }

  const int lane = tid & 31, w = tid >> 5;
  const int m_w = (w & 3) * 32, n_w = (w >> 2) * 64;
  const int cl = lane & 15, kh = lane >> 4;

  const int arow = tid >> 1, acs = (tid & 1) * 16;
  const size_t abase = (size_t)(u0 + arow) * DIM;
  const int bn = tid >> 1, bcs = (tid & 1) * 16;
  const size_t bbase = (size_t)(n0 + bn) * DIM;

  auto stageA = [&](int buf, int k0) {               // fused BN affine + relu
    const float4* p = (const float4*)(h1 + abase + k0 + acs);
    #pragma unroll
    for (int q = 0; q < 2; ++q) {
      float4 v0 = p[2*q], v1 = p[2*q+1];
      float vv[8] = {v0.x, v0.y, v0.z, v0.w, v1.x, v1.y, v1.z, v1.w};
      float r[8];
      #pragma unroll
      for (int c = 0; c < 8; ++c) {
        int g = k0 + acs + 8*q + c;
        float t = fmaf(vv[c], sScale[g], sShift[g]);
        r[c] = t > 0.0f ? t : 0.0f;
      }
      *(v8bf*)(&sA[buf][arow][acs + 8*q]) = pack8(r[0],r[1],r[2],r[3],r[4],r[5],r[6],r[7]);
    }
  };
  auto stageB = [&](int buf, int k0) {
    const float4* p = (const float4*)(W2 + bbase + k0 + bcs);
    #pragma unroll
    for (int q = 0; q < 2; ++q) {
      *(v8bf*)(&sBt[buf][bn][bcs + 8*q]) = pack8f4(p[2*q], p[2*q+1]);
    }
  };

  __syncthreads();                                   // sScale/sShift ready
  v8f acc[2][4] = {};
  int buf = 0;
  stageA(0, 0); stageB(0, 0);
  __syncthreads();
  for (int s = 0; s < NSTEPS_G; ++s) {
    if (s + 1 < NSTEPS_G) { stageA(buf ^ 1, (s + 1) * BK); stageB(buf ^ 1, (s + 1) * BK); }
    compute_step(sA[buf], sBt[buf], m_w, n_w, cl, kh, acc);
    __syncthreads();
    buf ^= 1;
  }

  #pragma unroll
  for (int i = 0; i < 2; ++i) {
    #pragma unroll
    for (int t = 0; t < 4; ++t) {
      int col = n0 + n_w + t * 16 + cl;
      #pragma unroll
      for (int v = 0; v < 8; ++v) {
        int r = m_w + i * 16 + kh * 8 + v;
        out[(size_t)(u0 + r) * DIM + col] = acc[i][t][v];
      }
    }
  }
}

// ---------------- host launcher ----------------
extern "C" void kernel_launch(void* const* d_in, const int* in_sizes, int n_in,
                              void* d_out, int out_size, void* d_ws, size_t ws_size,
                              hipStream_t stream) {
  const float* x_src  = (const float*)d_in[0];
  const int*   uniq   = (const int*)d_in[1];
  const int*   srcg   = (const int*)d_in[2];
  const float* tgt    = (const float*)d_in[3];
  const float* ef     = (const float*)d_in[4];
  const float* W_edge = (const float*)d_in[5];
  const float* b_edge = (const float*)d_in[6];
  const float* W1     = (const float*)d_in[7];
  const float* W2     = (const float*)d_in[8];
  const float* gamma  = (const float*)d_in[9];
  const float* beta   = (const float*)d_in[10];
  const float* eps    = (const float*)d_in[11];

  float* agg      = (float*)d_ws;                    // [U, D]
  float* h1       = agg + (size_t)U_NODES * DIM;     // [U, D]
  float* colsum   = h1  + (size_t)U_NODES * DIM;     // [D]
  float* colsumsq = colsum + DIM;
  float* scale    = colsumsq + DIM;
  float* shift    = scale + DIM;
  int*   idx      = (int*)(shift + DIM);             // [E]

  const int aggN = U_NODES * DIM;
  zero_f32_kernel<<<(aggN + 255) / 256, 256, 0, stream>>>(agg, aggN);
  zero_f32_kernel<<<(2 * DIM + 255) / 256, 256, 0, stream>>>(colsum, 2 * DIM);
  build_idx_kernel<<<(E_EDGES + 255) / 256, 256, 0, stream>>>(uniq, srcg, idx, E_EDGES, U_NODES);

  dim3 gridE(E_EDGES / E_BM, DIM / E_BN);   // 3125 x 3
  edge_gemm_kernel<<<gridE, NTHREADS, 0, stream>>>(tgt, ef, W_edge, b_edge, idx, agg);

  dim3 gridU(U_NODES / G_BM, DIM / G_BN);   // 64 x 6
  gemm1_kernel<<<gridU, NTHREADS, 0, stream>>>(x_src, agg, eps, W1, h1, colsum, colsumsq);
  bnstat_kernel<<<(DIM + 255) / 256, 256, 0, stream>>>(colsum, colsumsq, gamma, beta, scale, shift);
  gemm2_kernel<<<gridU, NTHREADS, 0, stream>>>(h1, scale, shift, W2, (float*)d_out);
}
